// MultiTaskTrunkNetwork_90658169684214
// MI455X (gfx1250) — matrix operand order repaired and checked by reading, not compiled, not run on hardware
//
#include <hip/hip_runtime.h>
#include <stdint.h>

typedef __attribute__((ext_vector_type(16))) __bf16 v16bf;
typedef __attribute__((ext_vector_type(8)))  float  v8f;

#define B_SZ   32768
#define IN_SZ  2056
#define H_SZ   1024
#define OUT_SZ 256
#define T_SZ   8
#define KPAD0  2080   // IN rounded up to multiple of 32 (zero padded)

#define BM 128
#define BN 128
#define BK 32
#define LDA 40        // LDS row stride in ushorts (80B) -> conflict-free frag reads
#define NTHREADS 256

// Dynamic-LDS byte layout (dynamic segment starts at offset 0; no static LDS in
// the GEMM kernels): A buffers at 0 and 10240, B buffers at 20480 and 30720,
// head row-index table at 40960.
#define ABUF_BYTES   (BM * LDA * 2)          // 10240
#define SB_BASE      (2 * ABUF_BYTES)        // 20480
#define SMEM_GEMM_BYTES (4 * ABUF_BYTES)     // 40960
#define SROWS_USHORT (SMEM_GEMM_BYTES / 2)   // ushort index of sRows

__device__ __forceinline__ unsigned int f2bf(float x) {
  unsigned int u = __float_as_uint(x);
  u += 0x7FFFu + ((u >> 16) & 1u);           // round-to-nearest-even
  return u >> 16;
}

__device__ __forceinline__ uint4 pack_bf16x8(float4 a, float4 b) {
  uint4 r;
  r.x = f2bf(a.x) | (f2bf(a.y) << 16);
  r.y = f2bf(a.z) | (f2bf(a.w) << 16);
  r.z = f2bf(b.x) | (f2bf(b.y) << 16);
  r.w = f2bf(b.z) | (f2bf(b.w) << 16);
  return r;
}

// ---- CDNA5 async global->LDS copy (ASYNCcnt tracked, no VGPR staging) ----
__device__ __forceinline__ void async_b128(unsigned lds_byte_off, const void* gaddr) {
  asm volatile("global_load_async_to_lds_b128 %0, %1, off"
               :: "v"(lds_byte_off), "v"((unsigned long long)(uintptr_t)gaddr)
               : "memory");
}
__device__ __forceinline__ void wait_async0() {
  asm volatile("s_wait_asynccnt 0x0" ::: "memory");
}

union FragU { uint4 q[2]; v16bf v; };

// A fragment (16x32, MxK): lane m=lane&15, kg=lane>>4; elems j<8 -> K=8kg+j,
// j>=8 -> K=16+8kg+(j-8). Two 16B LDS reads.
__device__ __forceinline__ v16bf load_frag_a(const unsigned short* buf, int mBase, int lane) {
  int m = lane & 15, kg = lane >> 4;
  const unsigned short* p = buf + (mBase + m) * LDA + kg * 8;
  FragU f;
  f.q[0] = *(const uint4*)(p);
  f.q[1] = *(const uint4*)(p + 16);
  return f.v;
}

// B fragment (32x16, KxN), weights stored transposed [n][k] in LDS:
// lane n=lane&15, kg=lane>>4; elems j<8 -> K=8kg+j, j>=8 -> K=16+8kg+(j-8).
__device__ __forceinline__ v16bf load_frag_b(const unsigned short* buf, int nBase, int lane) {
  int n = lane & 15, kg = lane >> 4;
  const unsigned short* p = buf + (nBase + n) * LDA + kg * 8;
  FragU f;
  f.q[0] = *(const uint4*)(p);
  f.q[1] = *(const uint4*)(p + 16);
  return f.v;
}

__device__ __forceinline__ v8f wmma_bf16(v16bf a, v16bf b, v8f c) {
  return __builtin_amdgcn_wmma_f32_16x16x32_bf16(false, a, false, b, (short)0, c, false, false);
}

// ---------------- input convert: X f32 [B][IN] -> Xb bf16 [B][KPAD0] ----------
__global__ __launch_bounds__(256) void convert_input(const float* __restrict__ X,
                                                     unsigned short* __restrict__ Xb) {
  int idx = blockIdx.x * blockDim.x + threadIdx.x;       // one per 8-col group
  int row = idx / (KPAD0 / 8);
  int g   = idx % (KPAD0 / 8);
  if (row >= B_SZ) return;
  uint4 o;
  if (g < IN_SZ / 8) {                                   // IN_SZ % 8 == 0
    const float4* p = (const float4*)(X + (size_t)row * IN_SZ + g * 8);
    o = pack_bf16x8(p[0], p[1]);
  } else {
    o = make_uint4(0, 0, 0, 0);
  }
  ((uint4*)(Xb + (size_t)row * KPAD0))[g] = o;
}

// ---------------- weight convert + transpose: W[K][N] f32 -> Wt[N][Kpad] bf16 --
__global__ __launch_bounds__(256) void transpose_w(const float* __restrict__ W,
                                                   unsigned short* __restrict__ Wt,
                                                   int K, int N, int Kpad) {
  __shared__ float tile[32][33];
  const float* Wz = W + (size_t)blockIdx.z * K * N;
  unsigned short* Wtz = Wt + (size_t)blockIdx.z * N * Kpad;
  int k0 = blockIdx.x * 32, n0 = blockIdx.y * 32;
  for (int i = threadIdx.y; i < 32; i += 8) {
    int k = k0 + i;
    tile[i][threadIdx.x] = (k < K) ? Wz[(size_t)k * N + n0 + threadIdx.x] : 0.0f;
  }
  __syncthreads();
  for (int i = threadIdx.y; i < 32; i += 8) {
    int n = n0 + i;
    Wtz[(size_t)n * Kpad + k0 + threadIdx.x] = (unsigned short)f2bf(tile[threadIdx.x][i]);
  }
}

// ---------------- task counting sort ----------------
__global__ void zero_counts(int* counts) { if (threadIdx.x < T_SZ) counts[threadIdx.x] = 0; }

__global__ void build_rowlist(const int* __restrict__ task, int* __restrict__ counts,
                              int* __restrict__ rowlist) {
  int i = blockIdx.x * blockDim.x + threadIdx.x;
  if (i < B_SZ) {
    int t = task[i];
    int pos = atomicAdd(&counts[t], 1);
    rowlist[t * B_SZ + pos] = i;
  }
}

// ---------------- trunk GEMM: Hout = bf16(tanh(A @ Wt^T + bias)) ----------------
// A: [M][Kpad] bf16, Bt: [N][Kpad] bf16 transposed weights. Fully async-fed.
__global__ __launch_bounds__(NTHREADS) void trunk_gemm(
    const unsigned short* __restrict__ A,
    const unsigned short* __restrict__ Bt,
    const float* __restrict__ bias,
    unsigned short* __restrict__ Hout,
    int Kpad, int Nout) {
  extern __shared__ unsigned short smem[];

  const int tid  = threadIdx.x;
  const int lane = tid & 31;
  const int wave = tid >> 5;
  const int wm   = wave & 3;     // 4 M-slabs of 32 rows
  const int wn   = wave >> 2;    // 2 N-slabs of 64 cols

  const int rowBase = blockIdx.y * BM;
  const int colBase = blockIdx.x * BN;

  const int ldRow  = tid >> 1;   // 0..127: one loader-pair per LDS row
  const int ldHalf = tid & 1;    // each thread feeds 32B per tile per matrix

  const int nKT = Kpad / BK;

  const unsigned short* aRowG = A  + (size_t)(rowBase + ldRow) * Kpad + ldHalf * 16;
  const unsigned short* bRowG = Bt + (size_t)(colBase + ldRow) * Kpad + ldHalf * 16;
  const unsigned tileOff = (unsigned)((ldRow * LDA + ldHalf * 16) * 2);  // byte off in buffer

#define ISSUE(kt_, buf_)                                                        \
  {                                                                             \
    const unsigned short* ga = aRowG + (kt_) * BK;                              \
    const unsigned short* gb = bRowG + (kt_) * BK;                              \
    unsigned la = (unsigned)((buf_) * ABUF_BYTES) + tileOff;                    \
    async_b128(la,      ga);                                                    \
    async_b128(la + 16, ga + 8);                                                \
    unsigned lb = (unsigned)(SB_BASE + (buf_) * ABUF_BYTES) + tileOff;          \
    async_b128(lb,      gb);                                                    \
    async_b128(lb + 16, gb + 8);                                                \
  }

  ISSUE(0, 0);
  wait_async0();
  __syncthreads();

  v8f acc[2][4] = {};

  for (int kt = 0; kt < nKT; ++kt) {
    const int buf = kt & 1;
    if (kt + 1 < nKT) ISSUE(kt + 1, buf ^ 1);  // async fill of other buffer

    const unsigned short* aBuf = smem + buf * (ABUF_BYTES / 2);
    const unsigned short* bBuf = smem + (SB_BASE / 2) + buf * (ABUF_BYTES / 2);

    v16bf af0 = load_frag_a(aBuf, wm * 32,      lane);
    v16bf af1 = load_frag_a(aBuf, wm * 32 + 16, lane);
    v16bf bf0 = load_frag_b(bBuf, wn * 64,      lane);
    v16bf bf1 = load_frag_b(bBuf, wn * 64 + 16, lane);
    v16bf bf2 = load_frag_b(bBuf, wn * 64 + 32, lane);
    v16bf bf3 = load_frag_b(bBuf, wn * 64 + 48, lane);

    acc[0][0] = wmma_bf16(af0, bf0, acc[0][0]);
    acc[0][1] = wmma_bf16(af0, bf1, acc[0][1]);
    acc[0][2] = wmma_bf16(af0, bf2, acc[0][2]);
    acc[0][3] = wmma_bf16(af0, bf3, acc[0][3]);
    acc[1][0] = wmma_bf16(af1, bf0, acc[1][0]);
    acc[1][1] = wmma_bf16(af1, bf1, acc[1][1]);
    acc[1][2] = wmma_bf16(af1, bf2, acc[1][2]);
    acc[1][3] = wmma_bf16(af1, bf3, acc[1][3]);

    if (kt + 1 < nKT) wait_async0();           // next buffer resident
    __syncthreads();                            // all waves done reading/filling
  }
#undef ISSUE

  const int kg = lane >> 4;
  const int nn = lane & 15;
  #pragma unroll
  for (int mi = 0; mi < 2; ++mi) {
    #pragma unroll
    for (int ni = 0; ni < 4; ++ni) {
      int col = colBase + wn * 64 + ni * 16 + nn;
      float bv = bias[col];
      #pragma unroll
      for (int r = 0; r < 8; ++r) {
        int row = rowBase + wm * 32 + mi * 16 + r + 8 * kg;  // C/D layout
        float v = tanhf(acc[mi][ni][r] + bv);
        Hout[(size_t)row * Nout + col] = (unsigned short)f2bf(v);
      }
    }
  }
}

// ---------------- per-task head GEMM with row gather/scatter ----------------
__global__ __launch_bounds__(NTHREADS) void head_gemm(
    const unsigned short* __restrict__ Hin,    // [B][H] bf16
    const unsigned short* __restrict__ WtH,    // [T][OUT][H] bf16
    const float* __restrict__ bH,              // [T][OUT]
    const int* __restrict__ counts,
    const int* __restrict__ rowlist,           // [T][B]
    float* __restrict__ Out) {                 // [B][OUT]
  const int t = blockIdx.z;
  const int cnt = counts[t];
  const int posBase = blockIdx.y * BM;
  if (posBase >= cnt) return;                  // uniform exit

  extern __shared__ unsigned short smem[];
  int* sRows = (int*)(smem + SROWS_USHORT);

  const int tid  = threadIdx.x;
  const int lane = tid & 31;
  const int wave = tid >> 5;
  const int wm   = wave & 3;
  const int wn   = wave >> 2;
  const int colBase = blockIdx.x * BN;
  const int ldRow  = tid >> 1;
  const int ldHalf = tid & 1;

  if (tid < BM) {
    int p = posBase + tid;
    sRows[tid] = (p < cnt) ? rowlist[t * B_SZ + p] : -1;
  }
  __syncthreads();

  const unsigned short* Bt = WtH + (size_t)t * OUT_SZ * H_SZ;
  const int nKT = H_SZ / BK;
  const int myRow = sRows[ldRow];              // constant across K loop
  const unsigned tileOff = (unsigned)((ldRow * LDA + ldHalf * 16) * 2);

  // Pre-zero A slots of invalid rows in BOTH buffers (async loads skip them).
  if (myRow < 0) {
    uint4 z = make_uint4(0, 0, 0, 0);
    #pragma unroll
    for (int b = 0; b < 2; ++b) {
      uint4* d = (uint4*)((char*)smem + b * ABUF_BYTES + tileOff);
      d[0] = z; d[1] = z;
    }
  }
  __syncthreads();

  const unsigned short* aRowG = (myRow >= 0)
      ? Hin + (size_t)myRow * H_SZ + ldHalf * 16 : (const unsigned short*)0;
  const unsigned short* bRowG = Bt + (size_t)(colBase + ldRow) * H_SZ + ldHalf * 16;

#define ISSUE_H(kt_, buf_)                                                      \
  {                                                                             \
    unsigned la = (unsigned)((buf_) * ABUF_BYTES) + tileOff;                    \
    if (myRow >= 0) {                                                           \
      const unsigned short* ga = aRowG + (kt_) * BK;                            \
      async_b128(la,      ga);                                                  \
      async_b128(la + 16, ga + 8);                                              \
    }                                                                           \
    const unsigned short* gb = bRowG + (kt_) * BK;                              \
    unsigned lb = (unsigned)(SB_BASE + (buf_) * ABUF_BYTES) + tileOff;          \
    async_b128(lb,      gb);                                                    \
    async_b128(lb + 16, gb + 8);                                                \
  }

  ISSUE_H(0, 0);
  wait_async0();
  __syncthreads();

  v8f acc[2][4] = {};
  for (int kt = 0; kt < nKT; ++kt) {
    const int buf = kt & 1;
    if (kt + 1 < nKT) ISSUE_H(kt + 1, buf ^ 1);

    const unsigned short* aBuf = smem + buf * (ABUF_BYTES / 2);
    const unsigned short* bBuf = smem + (SB_BASE / 2) + buf * (ABUF_BYTES / 2);

    v16bf af0 = load_frag_a(aBuf, wm * 32,      lane);
    v16bf af1 = load_frag_a(aBuf, wm * 32 + 16, lane);
    v16bf bf0 = load_frag_b(bBuf, wn * 64,      lane);
    v16bf bf1 = load_frag_b(bBuf, wn * 64 + 16, lane);
    v16bf bf2 = load_frag_b(bBuf, wn * 64 + 32, lane);
    v16bf bf3 = load_frag_b(bBuf, wn * 64 + 48, lane);

    acc[0][0] = wmma_bf16(af0, bf0, acc[0][0]);
    acc[0][1] = wmma_bf16(af0, bf1, acc[0][1]);
    acc[0][2] = wmma_bf16(af0, bf2, acc[0][2]);
    acc[0][3] = wmma_bf16(af0, bf3, acc[0][3]);
    acc[1][0] = wmma_bf16(af1, bf0, acc[1][0]);
    acc[1][1] = wmma_bf16(af1, bf1, acc[1][1]);
    acc[1][2] = wmma_bf16(af1, bf2, acc[1][2]);
    acc[1][3] = wmma_bf16(af1, bf3, acc[1][3]);

    if (kt + 1 < nKT) wait_async0();
    __syncthreads();
  }
#undef ISSUE_H

  const int kg = lane >> 4;
  const int nn = lane & 15;
  #pragma unroll
  for (int mi = 0; mi < 2; ++mi) {
    #pragma unroll
    for (int ni = 0; ni < 4; ++ni) {
      int col = colBase + wn * 64 + ni * 16 + nn;
      float bv = bH[t * OUT_SZ + col];
      #pragma unroll
      for (int r = 0; r < 8; ++r) {
        int loc = wm * 32 + mi * 16 + r + 8 * kg;
        if (posBase + loc < cnt) {
          int row = sRows[loc];
          Out[(size_t)row * OUT_SZ + col] = acc[mi][ni][r] + bv;
        }
      }
    }
  }
}

extern "C" void kernel_launch(void* const* d_in, const int* in_sizes, int n_in,
                              void* d_out, int out_size, void* d_ws, size_t ws_size,
                              hipStream_t stream) {
  (void)in_sizes; (void)n_in; (void)out_size; (void)ws_size;
  const float* inputs = (const float*)d_in[0];
  const int*   task   = (const int*)d_in[1];
  const float* W0 = (const float*)d_in[2];
  const float* b0 = (const float*)d_in[3];
  const float* W1 = (const float*)d_in[4];
  const float* b1 = (const float*)d_in[5];
  const float* W2 = (const float*)d_in[6];
  const float* b2 = (const float*)d_in[7];
  const float* Wh = (const float*)d_in[8];
  const float* bh = (const float*)d_in[9];

  char* ws = (char*)d_ws;
  size_t off = 0;
  auto alloc = [&](size_t bytes) {
    char* p = ws + off;
    off += (bytes + 255) & ~(size_t)255;
    return p;
  };
  unsigned short* Wt0 = (unsigned short*)alloc((size_t)H_SZ * KPAD0 * 2);
  unsigned short* Wt1 = (unsigned short*)alloc((size_t)H_SZ * H_SZ * 2);
  unsigned short* Wt2 = (unsigned short*)alloc((size_t)H_SZ * H_SZ * 2);
  unsigned short* WtH = (unsigned short*)alloc((size_t)T_SZ * OUT_SZ * H_SZ * 2);
  unsigned short* Xb  = (unsigned short*)alloc((size_t)B_SZ * KPAD0 * 2);  // also h1
  unsigned short* h0  = (unsigned short*)alloc((size_t)B_SZ * H_SZ * 2);
  int* counts  = (int*)alloc(T_SZ * sizeof(int));
  int* rowlist = (int*)alloc((size_t)T_SZ * B_SZ * sizeof(int));
  unsigned short* h1 = Xb;   // Xb is dead after layer 0; reuse as h1

  dim3 tb(32, 8);
  transpose_w<<<dim3(KPAD0 / 32, H_SZ / 32, 1), tb, 0, stream>>>(W0, Wt0, IN_SZ, H_SZ, KPAD0);
  transpose_w<<<dim3(H_SZ / 32, H_SZ / 32, 1), tb, 0, stream>>>(W1, Wt1, H_SZ, H_SZ, H_SZ);
  transpose_w<<<dim3(H_SZ / 32, H_SZ / 32, 1), tb, 0, stream>>>(W2, Wt2, H_SZ, H_SZ, H_SZ);
  transpose_w<<<dim3(H_SZ / 32, OUT_SZ / 32, T_SZ), tb, 0, stream>>>(Wh, WtH, H_SZ, OUT_SZ, H_SZ);

  convert_input<<<(B_SZ * (KPAD0 / 8)) / 256, 256, 0, stream>>>(inputs, Xb);

  zero_counts<<<1, 32, 0, stream>>>(counts);
  build_rowlist<<<B_SZ / 256, 256, 0, stream>>>(task, counts, rowlist);

  trunk_gemm<<<dim3(H_SZ / BN, B_SZ / BM), NTHREADS, SMEM_GEMM_BYTES, stream>>>(
      Xb, Wt0, b0, h0, KPAD0, H_SZ);
  trunk_gemm<<<dim3(H_SZ / BN, B_SZ / BM), NTHREADS, SMEM_GEMM_BYTES, stream>>>(
      h0, Wt1, b1, h1, H_SZ, H_SZ);
  trunk_gemm<<<dim3(H_SZ / BN, B_SZ / BM), NTHREADS, SMEM_GEMM_BYTES, stream>>>(
      h1, Wt2, b2, h0, H_SZ, H_SZ);

  head_gemm<<<dim3(OUT_SZ / BN, B_SZ / BM, T_SZ), NTHREADS, SMEM_GEMM_BYTES + 512, stream>>>(
      h0, WtH, bh, counts, rowlist, (float*)d_out);
}
